// MultiHeadAttention_66314295050804
// MI455X (gfx1250) — compile-verified, matrix operand order checked by reading, not run on hardware
//
#include <hip/hip_runtime.h>
#include <hip/hip_bf16.h>
#include <stdint.h>

typedef __attribute__((ext_vector_type(16))) __bf16 v16bf;
typedef __attribute__((ext_vector_type(8)))  float  v8f;

#define D_  1024
#define C_  2048
#define NH_ 16
#define H_  64
#define B_  2

union ABFrag { v16bf v; uint32_t u[8]; };

__device__ __forceinline__ uint16_t f2bf(float f) {
  union { float f; uint32_t u; } x; x.f = f;
  uint32_t r = x.u + 0x7fffu + ((x.u >> 16) & 1u);   // round-to-nearest-even
  return (uint16_t)(r >> 16);
}

// K-index of the first element of bf16 pair v (0..7) for a lane-half (A & B frag layouts)
__device__ __forceinline__ int pairK(int v, int half) {
  return ((v & 3) << 1) + ((v >> 2) << 4) + (half << 3);
}

__device__ __forceinline__ v8f wmma_bf16(const ABFrag& a, const ABFrag& b, v8f c) {
  return __builtin_amdgcn_wmma_f32_16x16x32_bf16(false, a.v, false, b.v,
                                                 (short)0, c, false, false);
}

// ---------------- Kernel 1: fp32 -> bf16 convert (vectorized x4) ----------------
__global__ void cvt_f32_bf16_v4(const float4* __restrict__ s, uint2* __restrict__ d, int n4) {
  int i = blockIdx.x * blockDim.x + threadIdx.x;
  if (i < n4) {
    float4 f = s[i];
    uint2 o;
    o.x = (uint32_t)f2bf(f.x) | ((uint32_t)f2bf(f.y) << 16);
    o.y = (uint32_t)f2bf(f.z) | ((uint32_t)f2bf(f.w) << 16);
    d[i] = o;
  }
}

// ---------------- Kernel 2: QKV projection GEMM ----------------
// M = B*C = 4096 rows of Xb[D], N = 3*D = 3072 features.
// Block: 128 thr = 4 waves; block tile 64(M) x 128(N); wave tile 64x32 (4x2 WMMA).
__global__ void __launch_bounds__(128) qkv_gemm(
    const uint16_t* __restrict__ Xb,   // [4096][1024]
    const uint16_t* __restrict__ Wqb,  // [1024][1024] (f-major, d contiguous)
    const uint16_t* __restrict__ Wkb,
    const uint16_t* __restrict__ Wvb,
    uint16_t* __restrict__ Qb,         // [B*NH][C][H]  (scaled by 1/8)
    uint16_t* __restrict__ Kb,         // [B*NH][C][H]
    uint16_t* __restrict__ Vtb)        // [B*NH][H][C]  (transposed)
{
  const int lane = threadIdx.x & 31;
  const int wave = threadIdx.x >> 5;
  const int half = lane >> 4;
  const int l15  = lane & 15;
  const int m0   = blockIdx.x * 64;
  const int f0   = blockIdx.y * 128 + wave * 32;   // global feature in [0,3072)
  const int mat  = f0 >> 10;                        // 0=Q 1=K 2=V (tile never straddles)
  const int fb   = f0 & 1023;
  const uint16_t* W = (mat == 0) ? Wqb : (mat == 1) ? Wkb : Wvb;

  v8f acc[4][2];
  for (int i = 0; i < 4; ++i) for (int j = 0; j < 2; ++j) acc[i][j] = (v8f){};

  for (int k0 = 0; k0 < D_; k0 += 32) {
    ABFrag a[4], b[2];
    for (int t = 0; t < 4; ++t) {
      const uint16_t* ap = Xb + (size_t)(m0 + t * 16 + l15) * D_ + k0;
      for (int v = 0; v < 8; ++v)
        a[t].u[v] = *(const uint32_t*)(ap + pairK(v, half));
    }
    for (int t = 0; t < 2; ++t) {
      const uint16_t* bp = W + (size_t)(fb + t * 16 + l15) * D_ + k0;
      for (int v = 0; v < 8; ++v)
        b[t].u[v] = *(const uint32_t*)(bp + pairK(v, half));
    }
    for (int i = 0; i < 4; ++i)
      for (int j = 0; j < 2; ++j)
        acc[i][j] = wmma_bf16(a[i], b[j], acc[i][j]);
  }

  for (int i = 0; i < 4; ++i)
    for (int j = 0; j < 2; ++j)
      for (int r = 0; r < 8; ++r) {
        int m = m0 + i * 16 + r + half * 8;
        int f = fb + j * 16 + l15;
        int bi = m / C_, c = m & (C_ - 1);
        int head = f >> 6, h = f & 63;
        float vv = acc[i][j][r];
        if (mat == 0)
          Qb[(size_t)((bi * NH_ + head) * C_ + c) * H_ + h] = f2bf(vv * 0.125f);
        else if (mat == 1)
          Kb[(size_t)((bi * NH_ + head) * C_ + c) * H_ + h] = f2bf(vv);
        else
          Vtb[(size_t)((bi * NH_ + head) * H_ + h) * C_ + c] = f2bf(vv);
      }
}

// ---------------- Kernel 3: causal flash attention ----------------
// grid.x = B*NH (32), grid.y = C/64 (32). 4 waves; wave owns 16 q rows.
__global__ void __launch_bounds__(128) attn(
    const uint16_t* __restrict__ Qb, const uint16_t* __restrict__ Kb,
    const uint16_t* __restrict__ Vtb, uint16_t* __restrict__ attO)  // [B][C][D]
{
  __shared__ __align__(16) uint16_t pbuf[4][16][32];  // per-wave P staging
  const int lane = threadIdx.x & 31;
  const int wave = threadIdx.x >> 5;
  const int half = lane >> 4;
  const int l15  = lane & 15;
  const int bh   = blockIdx.x;
  const int q0b  = blockIdx.y * 64;
  const int q0   = q0b + wave * 16;
  const size_t qkBase = (size_t)bh * C_ * H_;
  const size_t vBase  = (size_t)bh * H_ * C_;

  ABFrag aq[2];
  {
    const uint16_t* qp = Qb + qkBase + (size_t)(q0 + l15) * H_;
    for (int t = 0; t < 2; ++t)
      for (int v = 0; v < 8; ++v)
        aq[t].u[v] = *(const uint32_t*)(qp + t * 32 + pairK(v, half));
  }

  v8f o[4];
  for (int nt = 0; nt < 4; ++nt) o[nt] = (v8f){};
  float mrow[8], lrow[8];
  for (int r = 0; r < 8; ++r) { mrow[r] = -1e30f; lrow[r] = 0.f; }

  const int nch = q0b / 32 + 2;  // causal: kv <= q0b+63
  for (int ch = 0; ch < nch; ++ch) {
    const int kc = ch * 32;
    // S = Q * K^T  (16 q x 32 kv), two 16x16 tiles, K-dim = H = 2x32
    v8f s[2];
    for (int n2 = 0; n2 < 2; ++n2) {
      ABFrag bk[2];
      const uint16_t* kp = Kb + qkBase + (size_t)(kc + n2 * 16 + l15) * H_;
      for (int t = 0; t < 2; ++t)
        for (int v = 0; v < 8; ++v)
          bk[t].u[v] = *(const uint32_t*)(kp + t * 32 + pairK(v, half));
      v8f z = (v8f){};
      z = wmma_bf16(aq[0], bk[0], z);
      s[n2] = wmma_bf16(aq[1], bk[1], z);
    }
    // causal mask + row max over 32 cols
    float sn[8];
    for (int r = 0; r < 8; ++r) {
      int q = q0 + r + half * 8;
      float v0 = (kc + l15 <= q)      ? s[0][r] : -1e30f;
      float v1 = (kc + 16 + l15 <= q) ? s[1][r] : -1e30f;
      s[0][r] = v0; s[1][r] = v1;
      sn[r] = fmaxf(v0, v1);
    }
    for (int msk = 1; msk < 16; msk <<= 1)
      for (int r = 0; r < 8; ++r)
        sn[r] = fmaxf(sn[r], __shfl_xor(sn[r], msk, 32));
    float alpha[8];
    for (int r = 0; r < 8; ++r) {
      float mn = fmaxf(mrow[r], sn[r]);
      alpha[r] = __expf(mrow[r] - mn);
      mrow[r] = mn;
    }
    float ps[8];
    for (int r = 0; r < 8; ++r) {
      float p0 = __expf(s[0][r] - mrow[r]);
      float p1 = __expf(s[1][r] - mrow[r]);
      pbuf[wave][r + half * 8][l15]      = f2bf(p0);
      pbuf[wave][r + half * 8][16 + l15] = f2bf(p1);
      ps[r] = p0 + p1;
    }
    for (int msk = 1; msk < 16; msk <<= 1)
      for (int r = 0; r < 8; ++r)
        ps[r] += __shfl_xor(ps[r], msk, 32);
    for (int r = 0; r < 8; ++r) lrow[r] = lrow[r] * alpha[r] + ps[r];
    for (int nt = 0; nt < 4; ++nt)
      for (int r = 0; r < 8; ++r) o[nt][r] *= alpha[r];
    __syncthreads();
    // P (C/D layout) -> A fragment via LDS
    ABFrag ap;
    for (int v = 0; v < 8; ++v)
      ap.u[v] = *(const uint32_t*)&pbuf[wave][l15][pairK(v, half)];
    // O += P * V   (V transposed: contiguous in kv per lane)
    for (int nt = 0; nt < 4; ++nt) {
      ABFrag bv;
      const uint16_t* vp = Vtb + vBase + (size_t)(nt * 16 + l15) * C_ + kc;
      for (int v = 0; v < 8; ++v) bv.u[v] = *(const uint32_t*)(vp + pairK(v, half));
      o[nt] = wmma_bf16(ap, bv, o[nt]);
    }
    __syncthreads();
  }

  const int head = bh & (NH_ - 1);
  const int bi   = bh >> 4;
  for (int r = 0; r < 8; ++r) lrow[r] = 1.f / lrow[r];
  for (int nt = 0; nt < 4; ++nt)
    for (int r = 0; r < 8; ++r) {
      int q = q0 + r + half * 8;
      int h = nt * 16 + l15;
      attO[(size_t)(bi * C_ + q) * D_ + head * H_ + h] = f2bf(o[nt][r] * lrow[r]);
    }
}

// ---------------- Kernel 4: output projection + bias + residual ----------------
// Block tile 64(M) x 128(N); wave tile 64x32 (4x2 WMMA).
__global__ void __launch_bounds__(128) proj_gemm(
    const uint16_t* __restrict__ Ab,   // attO bf16 [4096][1024]
    const uint16_t* __restrict__ Wpb,  // [1024][1024] (i-major, j contiguous)
    const float* __restrict__ bp, const float* __restrict__ X,
    float* __restrict__ out)
{
  const int lane = threadIdx.x & 31;
  const int wave = threadIdx.x >> 5;
  const int half = lane >> 4;
  const int l15  = lane & 15;
  const int m0 = blockIdx.x * 64;
  const int n0 = blockIdx.y * 128 + wave * 32;

  v8f acc[4][2];
  for (int i = 0; i < 4; ++i) for (int j = 0; j < 2; ++j) acc[i][j] = (v8f){};

  for (int k0 = 0; k0 < D_; k0 += 32) {
    ABFrag a[4], b[2];
    for (int t = 0; t < 4; ++t) {
      const uint16_t* ap = Ab + (size_t)(m0 + t * 16 + l15) * D_ + k0;
      for (int v = 0; v < 8; ++v)
        a[t].u[v] = *(const uint32_t*)(ap + pairK(v, half));
    }
    for (int t = 0; t < 2; ++t) {
      const uint16_t* bq = Wpb + (size_t)(n0 + t * 16 + l15) * D_ + k0;
      for (int v = 0; v < 8; ++v)
        b[t].u[v] = *(const uint32_t*)(bq + pairK(v, half));
    }
    for (int i = 0; i < 4; ++i)
      for (int j = 0; j < 2; ++j)
        acc[i][j] = wmma_bf16(a[i], b[j], acc[i][j]);
  }

  for (int i = 0; i < 4; ++i)
    for (int j = 0; j < 2; ++j)
      for (int r = 0; r < 8; ++r) {
        int m = m0 + i * 16 + r + half * 8;
        int n = n0 + j * 16 + l15;
        out[(size_t)m * D_ + n] = acc[i][j][r] + bp[n] + X[(size_t)m * D_ + n];
      }
}

// ---------------- launcher ----------------
extern "C" void kernel_launch(void* const* d_in, const int* in_sizes, int n_in,
                              void* d_out, int out_size, void* d_ws, size_t ws_size,
                              hipStream_t stream) {
  const float* X  = (const float*)d_in[0];
  const float* Wq = (const float*)d_in[1];
  const float* Wk = (const float*)d_in[2];
  const float* Wv = (const float*)d_in[3];
  const float* Wp = (const float*)d_in[4];
  const float* bp = (const float*)d_in[5];
  float* out = (float*)d_out;

  const size_t MB = 1024 * 1024;
  char* ws = (char*)d_ws;
  uint16_t* Xb   = (uint16_t*)(ws + 0);        // 8 MB  (4M el)
  uint16_t* Wqb  = (uint16_t*)(ws + 8  * MB);  // 2 MB
  uint16_t* Wkb  = (uint16_t*)(ws + 10 * MB);  // 2 MB
  uint16_t* Wvb  = (uint16_t*)(ws + 12 * MB);  // 2 MB
  uint16_t* Wpb  = (uint16_t*)(ws + 14 * MB);  // 2 MB
  uint16_t* Qb   = (uint16_t*)(ws + 16 * MB);  // 8 MB
  uint16_t* Kb   = (uint16_t*)(ws + 24 * MB);  // 8 MB
  uint16_t* Vtb  = (uint16_t*)(ws + 32 * MB);  // 8 MB
  uint16_t* attO = (uint16_t*)(ws + 40 * MB);  // 8 MB

  const int nX4 = (B_ * C_ * D_) / 4;   // 1048576 float4
  const int nW4 = (NH_ * H_ * D_) / 4;  // 262144 float4
  cvt_f32_bf16_v4<<<(nX4 + 255) / 256, 256, 0, stream>>>((const float4*)X,  (uint2*)Xb,  nX4);
  cvt_f32_bf16_v4<<<(nW4 + 255) / 256, 256, 0, stream>>>((const float4*)Wq, (uint2*)Wqb, nW4);
  cvt_f32_bf16_v4<<<(nW4 + 255) / 256, 256, 0, stream>>>((const float4*)Wk, (uint2*)Wkb, nW4);
  cvt_f32_bf16_v4<<<(nW4 + 255) / 256, 256, 0, stream>>>((const float4*)Wv, (uint2*)Wvb, nW4);
  cvt_f32_bf16_v4<<<(nW4 + 255) / 256, 256, 0, stream>>>((const float4*)Wp, (uint2*)Wpb, nW4);

  qkv_gemm<<<dim3((B_ * C_) / 64, (3 * D_) / 128), 128, 0, stream>>>(
      Xb, Wqb, Wkb, Wvb, Qb, Kb, Vtb);

  attn<<<dim3(B_ * NH_, C_ / 64), 128, 0, stream>>>(Qb, Kb, Vtb, attO);

  proj_gemm<<<dim3((B_ * C_) / 64, D_ / 128), 128, 0, stream>>>(
      attO, Wpb, bp, X, out);
}
